// GATTransformerNodeOnly_86088324481819
// MI455X (gfx1250) — compile-verified, matrix operand order checked by reading, not run on hardware
//
#include <hip/hip_runtime.h>
#include <hip/hip_bf16.h>

// ---------------------------------------------------------------------------
// Problem constants (fixed by the reference's setup_inputs)
// ---------------------------------------------------------------------------
#define NNODES 8192      // B * NPG
#define NEDGES 131072
#define HIDC   256
#define NLAY   4         // GAT layers
#define TLAY   2         // transformer layers
#define NB     8         // graphs
#define SEQ    1024      // nodes per graph
#define NHEAD  8         // transformer heads
#define HD     32        // transformer head dim
#define GHEAD  4         // GAT heads
#define GDH    64        // GAT head dim
#define EHC    32        // edge feature dim

typedef __bf16 v16bf __attribute__((ext_vector_type(16)));
typedef float  v8f   __attribute__((ext_vector_type(8)));

union Frag16 { v16bf v; unsigned short s[16]; uint4 q[2]; };

__device__ __forceinline__ unsigned short f2bf(float f) {
  unsigned u = __float_as_uint(f);
  unsigned r = (u + 0x7fffu + ((u >> 16) & 1u)) >> 16;   // round-to-nearest-even
  return (unsigned short)r;
}

__device__ __forceinline__ float gelu_erf(float x) {
  return 0.5f * x * (1.0f + erff(x * 0.70710678118654752f));
}

// order-preserving float->uint encoding for atomicMax-based segment max
__device__ __forceinline__ unsigned enc_f(float x) {
  unsigned u = __float_as_uint(x);
  return (u & 0x80000000u) ? ~u : (u | 0x80000000u);
}
__device__ __forceinline__ float dec_f(unsigned u) {
  return __uint_as_float((u & 0x80000000u) ? (u & 0x7fffffffu) : ~u);
}

// ---------------------------------------------------------------------------
// Pack a K x N f32 weight matrix into WMMA B-fragment order (bf16):
// fragment (kc, nt) = 32x16 tile (K rows kc*32.., cols nt*16..); each fragment
// is 32 lanes x 16 bf16 contiguous (1KB). One thread per (fragment, lane).
// Fragment-lane element layout matches ISA 7.12.2 16-bit B operand:
//   lane L: col = nt*16 + (L&15); elems j   -> K = kc*32 + (L>>4)*8 + j
//                                 elems 8+j -> K = kc*32 + 16 + (L>>4)*8 + j
// ---------------------------------------------------------------------------
__global__ void pack_w_kernel(const float* __restrict__ W,
                              unsigned short* __restrict__ Wp,
                              int N, int K)
{
  int gid = blockIdx.x * 256 + threadIdx.x;
  int total = (K >> 5) * (N >> 4) * 32;
  if (gid >= total) return;
  int lane = gid & 31;
  int frag = gid >> 5;
  int ntiles = N >> 4;
  int kc = frag / ntiles;
  int nt = frag - kc * ntiles;
  int r = lane & 15, kh = lane >> 4;
  int col = nt * 16 + r;
  const float* w0 = W + (size_t)(kc * 32 + kh * 8) * N + col;
  Frag16 t;
#pragma unroll
  for (int j = 0; j < 8; ++j) {
    t.s[j]     = f2bf(w0[(size_t)j * N]);
    t.s[8 + j] = f2bf(w0[(size_t)(16 + j) * N]);
  }
  uint4* out = (uint4*)(Wp + (size_t)frag * 512 + lane * 16);
  out[0] = t.q[0];
  out[1] = t.q[1];
}

// ---------------------------------------------------------------------------
// Generic WMMA GEMM: C[M,N] = A[M,K](bf16 row-major) @ W(packed frags)
// one wave per 16x64 tile of C; 4 WMMAs per 32-deep K step.
// act: 0 = none, 1 = erf-GELU.  N % 64 == 0, K % 32 == 0, M % 16 == 0.
// ---------------------------------------------------------------------------
__global__ void wmma_gemm_kernel(const unsigned short* __restrict__ A,
                                 const unsigned short* __restrict__ Wp,
                                 const float* __restrict__ bias,
                                 const float* __restrict__ res,
                                 float* __restrict__ C,
                                 int M, int N, int K, int act)
{
  const int tn = blockIdx.x * 64;
  const int tm = blockIdx.y * 16;
  const int lane = threadIdx.x;
  const int r  = lane & 15;
  const int kh = lane >> 4;
  const int ntiles = N >> 4;

  v8f acc0 = {}, acc1 = {}, acc2 = {}, acc3 = {};
  const unsigned short* arow = A + (size_t)(tm + r) * K;

  for (int k0 = 0; k0 < K; k0 += 32) {
    Frag16 af;
    {
      const unsigned short* ap = arow + k0 + kh * 8;
      af.q[0] = *(const uint4*)(ap);
      af.q[1] = *(const uint4*)(ap + 16);
      if (k0 + 32 < K) __builtin_prefetch((const void*)(ap + 32), 0, 3);
    }
    const unsigned short* bbase =
        Wp + ((size_t)(k0 >> 5) * ntiles + (tn >> 4)) * 512 + lane * 16;
    Frag16 b0, b1, b2, b3;
    b0.q[0] = *(const uint4*)(bbase);         b0.q[1] = *(const uint4*)(bbase + 8);
    b1.q[0] = *(const uint4*)(bbase + 512);   b1.q[1] = *(const uint4*)(bbase + 520);
    b2.q[0] = *(const uint4*)(bbase + 1024);  b2.q[1] = *(const uint4*)(bbase + 1032);
    b3.q[0] = *(const uint4*)(bbase + 1536);  b3.q[1] = *(const uint4*)(bbase + 1544);
    acc0 = __builtin_amdgcn_wmma_f32_16x16x32_bf16(false, af.v, false, b0.v, (short)0, acc0, false, false);
    acc1 = __builtin_amdgcn_wmma_f32_16x16x32_bf16(false, af.v, false, b1.v, (short)0, acc1, false, false);
    acc2 = __builtin_amdgcn_wmma_f32_16x16x32_bf16(false, af.v, false, b2.v, (short)0, acc2, false, false);
    acc3 = __builtin_amdgcn_wmma_f32_16x16x32_bf16(false, af.v, false, b3.v, (short)0, acc3, false, false);
  }

  const int rowbase = kh * 8;
  v8f accs[4] = {acc0, acc1, acc2, acc3};
#pragma unroll
  for (int t = 0; t < 4; ++t) {
#pragma unroll
    for (int i = 0; i < 8; ++i) {
      int row = tm + rowbase + i;
      int cc  = tn + t * 16 + r;
      float v = accs[t][i];
      if (bias) v += bias[cc];
      if (act == 1) v = gelu_erf(v);
      if (res) v += res[(size_t)row * N + cc];
      C[(size_t)row * N + cc] = v;
    }
  }
}

// ---------------------------------------------------------------------------
// f32 -> bf16 conversion
// ---------------------------------------------------------------------------
__global__ void cvt_bf16_kernel(const float* __restrict__ in,
                                unsigned short* __restrict__ out, int n)
{
  int i = blockIdx.x * 256 + threadIdx.x;
  if (i < n) out[i] = f2bf(in[i]);
}

// ---------------------------------------------------------------------------
// Pack V [N,256] f32 into attention B-fragment order (bf16):
// fragment = ((b*NHEAD+head)*(SEQ/32)+chunk)*2 + f, where f selects head-dim
// half (dims f*16..f*16+15). One thread per (fragment, lane).
// ---------------------------------------------------------------------------
__global__ void pack_v_kernel(const float* __restrict__ v,
                              unsigned short* __restrict__ vp)
{
  int gid = blockIdx.x * 256 + threadIdx.x;          // 4096 frags * 32 lanes
  if (gid >= NB * NHEAD * (SEQ / 32) * 2 * 32) return;
  int lane = gid & 31;
  int frag = gid >> 5;
  int f     = frag & 1;
  int chunk = (frag >> 1) & 31;
  int head  = (frag >> 6) & 7;
  int b     = frag >> 9;
  int r = lane & 15, kh = lane >> 4;
  size_t base = (size_t)b * SEQ;
  int d = head * HD + f * 16 + r;
  const float* v0 = v + (base + chunk * 32 + kh * 8) * HIDC + d;
  Frag16 t;
#pragma unroll
  for (int j = 0; j < 8; ++j) {
    t.s[j]     = f2bf(v0[(size_t)j * HIDC]);
    t.s[8 + j] = f2bf(v0[(size_t)(16 + j) * HIDC]);
  }
  uint4* out = (uint4*)(vp + (size_t)frag * 512 + lane * 16);
  out[0] = t.q[0];
  out[1] = t.q[1];
}

// ---------------------------------------------------------------------------
// Fused node encoder: h = gelu(x@w1+b1)@w2+b2   (6 -> 64 -> 256), wave/node
// ---------------------------------------------------------------------------
__global__ void node_enc_kernel(const float* __restrict__ x,
                                const float* __restrict__ w1, const float* __restrict__ b1,
                                const float* __restrict__ w2, const float* __restrict__ b2,
                                float* __restrict__ h, int n_rows)
{
  __shared__ float hid[8][64];
  int wave = threadIdx.x >> 5, lane = threadIdx.x & 31;
  int n = blockIdx.x * 8 + wave;
  if (n < n_rows) {
#pragma unroll
    for (int t = 0; t < 2; ++t) {
      int j = lane * 2 + t;
      float s = b1[j];
#pragma unroll
      for (int i = 0; i < 6; ++i) s += x[(size_t)n * 6 + i] * w1[i * 64 + j];
      hid[wave][j] = gelu_erf(s);
    }
  }
  __syncthreads();
  if (n < n_rows) {
#pragma unroll
    for (int t = 0; t < 8; ++t) {
      int c = t * 32 + lane;
      float s = b2[c];
      for (int j = 0; j < 64; ++j) s += hid[wave][j] * w2[j * HIDC + c];
      h[(size_t)n * HIDC + c] = s;
    }
  }
}

// ---------------------------------------------------------------------------
// Fused edge encoder: e = gelu(ea@w1+b1)@w2+b2   (4 -> 32 -> 32), wave/edge
// ---------------------------------------------------------------------------
__global__ void edge_enc_kernel(const float* __restrict__ ea,
                                const float* __restrict__ w1, const float* __restrict__ b1,
                                const float* __restrict__ w2, const float* __restrict__ b2,
                                float* __restrict__ e, int n_edges)
{
  __shared__ float hid[8][32];
  int wave = threadIdx.x >> 5, lane = threadIdx.x & 31;
  int ed = blockIdx.x * 8 + wave;
  if (ed < n_edges) {
    float s = b1[lane];
#pragma unroll
    for (int i = 0; i < 4; ++i) s += ea[(size_t)ed * 4 + i] * w1[i * 32 + lane];
    hid[wave][lane] = gelu_erf(s);
  }
  __syncthreads();
  if (ed < n_edges) {
    float s = b2[lane];
    for (int j = 0; j < 32; ++j) s += hid[wave][j] * w2[j * 32 + lane];
    e[(size_t)ed * 32 + lane] = s;
  }
}

// ---------------------------------------------------------------------------
// GAT pass 1: per-edge logits (ee computed on the fly) + segment max
// one wave per edge; lane handles 8 channels; 8-lane groups = one head
// ---------------------------------------------------------------------------
__global__ void gat_logit_kernel(const float* __restrict__ e,
                                 const float* __restrict__ we,   // [32,256]
                                 const float* __restrict__ xl,
                                 const float* __restrict__ xr,
                                 const int* __restrict__ src,
                                 const int* __restrict__ dst,
                                 const float* __restrict__ att,  // [256]
                                 float* __restrict__ logit,      // [E,4]
                                 unsigned* __restrict__ menc,    // [N,4]
                                 int n_edges)
{
  int wave = threadIdx.x >> 5, lane = threadIdx.x & 31;
  int ed = blockIdx.x * 8 + wave;
  if (ed >= n_edges) return;
  int s = src[ed], d = dst[ed];
  float eval = e[(size_t)ed * EHC + lane];
  float acc = 0.f;
#pragma unroll
  for (int i = 0; i < 8; ++i) {
    int c = lane * 8 + i;
    float ee = 0.f;
    for (int j = 0; j < EHC; ++j) ee += __shfl(eval, j, 32) * we[j * HIDC + c];
    float z = xl[(size_t)d * HIDC + c] + xr[(size_t)s * HIDC + c] + ee;
    z = z > 0.f ? z : 0.2f * z;              // leaky_relu 0.2
    acc += z * att[c];
  }
  acc += __shfl_xor(acc, 1, 32);
  acc += __shfl_xor(acc, 2, 32);
  acc += __shfl_xor(acc, 4, 32);
  if ((lane & 7) == 0) {
    int head = lane >> 3;
    logit[(size_t)ed * GHEAD + head] = acc;
    atomicMax(&menc[(size_t)d * GHEAD + head], enc_f(acc));
  }
}

// GAT pass 2: ex = exp(logit - m[dst]); den[dst] += ex
__global__ void gat_softmax_kernel(const float* __restrict__ logit,
                                   const unsigned* __restrict__ menc,
                                   const int* __restrict__ dst,
                                   float* __restrict__ exbuf,
                                   float* __restrict__ den, int n_items)
{
  int i = blockIdx.x * 256 + threadIdx.x;
  if (i >= n_items) return;
  int ed = i >> 2, head = i & 3;
  int d = dst[ed];
  float m = dec_f(menc[(size_t)d * GHEAD + head]);
  float ex = __expf(logit[i] - m);
  exbuf[i] = ex;
  atomicAdd(&den[(size_t)d * GHEAD + head], ex);
}

// GAT pass 3: agg[dst] += xr[src] * alpha  (one block of 256 per edge)
__global__ void gat_scatter_kernel(const float* __restrict__ xr,
                                   const float* __restrict__ exbuf,
                                   const float* __restrict__ den,
                                   const int* __restrict__ src,
                                   const int* __restrict__ dst,
                                   float* __restrict__ agg, int n_edges)
{
  int ed = blockIdx.x;
  if (ed >= n_edges) return;
  int c = threadIdx.x;
  int s = src[ed], d = dst[ed];
  int head = c >> 6;
  float alpha = exbuf[(size_t)ed * GHEAD + head] /
                (den[(size_t)d * GHEAD + head] + 1e-16f);
  atomicAdd(&agg[(size_t)d * HIDC + c], xr[(size_t)s * HIDC + c] * alpha);
}

// ---------------------------------------------------------------------------
// LayerNorm over 256 channels, one wave per row:
// out = LN(x (+add) (+addb)) * g + b
// ---------------------------------------------------------------------------
__global__ void ln_kernel(const float* __restrict__ x,
                          const float* __restrict__ add,
                          const float* __restrict__ addb,
                          const float* __restrict__ g,
                          const float* __restrict__ b,
                          float* __restrict__ out, int n_rows)
{
  int wave = threadIdx.x >> 5, lane = threadIdx.x & 31;
  int row = blockIdx.x * 8 + wave;
  if (row >= n_rows) return;
  float t[8];
  float s = 0.f;
#pragma unroll
  for (int i = 0; i < 8; ++i) {
    int c = i * 32 + lane;
    float v = x[(size_t)row * HIDC + c];
    if (add)  v += add[(size_t)row * HIDC + c];
    if (addb) v += addb[c];
    t[i] = v; s += v;
  }
#pragma unroll
  for (int off = 16; off > 0; off >>= 1) s += __shfl_xor(s, off, 32);
  float mean = s * (1.0f / HIDC);
  float s2 = 0.f;
#pragma unroll
  for (int i = 0; i < 8; ++i) { float dv = t[i] - mean; s2 += dv * dv; }
#pragma unroll
  for (int off = 16; off > 0; off >>= 1) s2 += __shfl_xor(s2, off, 32);
  float rstd = rsqrtf(s2 * (1.0f / HIDC) + 1e-5f);
#pragma unroll
  for (int i = 0; i < 8; ++i) {
    int c = i * 32 + lane;
    out[(size_t)row * HIDC + c] = (t[i] - mean) * rstd * g[c] + b[c];
  }
}

// ---------------------------------------------------------------------------
// Flash-attention tile: one wave per (graph, head, 16-query tile).
// scores via 2x WMMA per 32-key chunk, online softmax through LDS,
// P x V via 2x accumulating WMMA (V pre-packed into B-fragment order).
// Mask is all-true (every graph is exactly SEQ nodes).
// ---------------------------------------------------------------------------
__global__ void attn_kernel(const unsigned short* __restrict__ qb,
                            const unsigned short* __restrict__ kb,
                            const unsigned short* __restrict__ vpk,
                            float* __restrict__ out)
{
  __shared__ float lds_s[16 * 32];
  __shared__ float lds_f[16];
  const int lane = threadIdx.x;
  const int r  = lane & 15;
  const int kh = lane >> 4;
  const int rowbase = kh * 8;
  const int q0   = blockIdx.x * 16;
  const int head = blockIdx.y;
  const int bg   = blockIdx.z;
  const size_t base = (size_t)bg * SEQ;
  const int hoff = head * HD;
  const float scale = 0.17677669529663687f;   // 1/sqrt(32)

  Frag16 qf;
  {
    const unsigned short* p = qb + (base + q0 + r) * HIDC + hoff + kh * 8;
    qf.q[0] = *(const uint4*)(p);
    qf.q[1] = *(const uint4*)(p + 16);
  }

  // packed V fragments for this (graph, head): [chunk][2][lane][16]
  const unsigned short* vbase =
      vpk + ((size_t)(bg * NHEAD + head) * (SEQ / 32) * 2) * 512 + lane * 16;

  v8f acc0 = {}, acc1 = {};
  float m_i = -1e30f, l_i = 0.0f;            // valid on lanes < 16 (row = lane)

  for (int kt = 0; kt < SEQ; kt += 32) {
    Frag16 kf0, kf1;
    {
      const unsigned short* p0 = kb + (base + kt + r)      * HIDC + hoff + kh * 8;
      const unsigned short* p1 = kb + (base + kt + 16 + r) * HIDC + hoff + kh * 8;
      kf0.q[0] = *(const uint4*)(p0);  kf0.q[1] = *(const uint4*)(p0 + 16);
      kf1.q[0] = *(const uint4*)(p1);  kf1.q[1] = *(const uint4*)(p1 + 16);
    }
    v8f s0 = {}, s1 = {};
    s0 = __builtin_amdgcn_wmma_f32_16x16x32_bf16(false, qf.v, false, kf0.v, (short)0, s0, false, false);
    s1 = __builtin_amdgcn_wmma_f32_16x16x32_bf16(false, qf.v, false, kf1.v, (short)0, s1, false, false);

    __syncthreads();                          // previous P reads done
#pragma unroll
    for (int i = 0; i < 8; ++i) {
      lds_s[(rowbase + i) * 32 + r]      = s0[i] * scale;
      lds_s[(rowbase + i) * 32 + 16 + r] = s1[i] * scale;
    }
    __syncthreads();
    if (lane < 16) {                          // lane owns row = lane
      float mx = -1e30f;
#pragma unroll
      for (int j = 0; j < 32; ++j) mx = fmaxf(mx, lds_s[lane * 32 + j]);
      float newm = fmaxf(m_i, mx);
      float corr = __expf(m_i - newm);
      float ssum = 0.f;
#pragma unroll
      for (int j = 0; j < 32; ++j) {
        float pv = __expf(lds_s[lane * 32 + j] - newm);
        lds_s[lane * 32 + j] = pv;
        ssum += pv;
      }
      l_i = l_i * corr + ssum;
      m_i = newm;
      lds_f[lane] = corr;
    }
    __syncthreads();
#pragma unroll
    for (int i = 0; i < 8; ++i) {
      float c = lds_f[rowbase + i];
      acc0[i] *= c; acc1[i] *= c;
    }
    Frag16 pf;
#pragma unroll
    for (int j = 0; j < 8; ++j) {
      pf.s[j]     = f2bf(lds_s[r * 32 + kh * 8 + j]);
      pf.s[8 + j] = f2bf(lds_s[r * 32 + 16 + kh * 8 + j]);
    }
    Frag16 vf0, vf1;
    {
      const unsigned short* pv = vbase + (size_t)(kt >> 5) * 1024;
      vf0.q[0] = *(const uint4*)(pv);       vf0.q[1] = *(const uint4*)(pv + 8);
      vf1.q[0] = *(const uint4*)(pv + 512); vf1.q[1] = *(const uint4*)(pv + 520);
    }
    acc0 = __builtin_amdgcn_wmma_f32_16x16x32_bf16(false, pf.v, false, vf0.v, (short)0, acc0, false, false);
    acc1 = __builtin_amdgcn_wmma_f32_16x16x32_bf16(false, pf.v, false, vf1.v, (short)0, acc1, false, false);
  }

  __syncthreads();
  if (lane < 16) lds_f[lane] = 1.0f / l_i;
  __syncthreads();
#pragma unroll
  for (int i = 0; i < 8; ++i) {
    float inv = lds_f[rowbase + i];
    size_t row = base + q0 + rowbase + i;
    out[row * HIDC + hoff + r]      = acc0[i] * inv;
    out[row * HIDC + hoff + 16 + r] = acc1[i] * inv;
  }
}

// ---------------------------------------------------------------------------
// Output head: out[n,3] = ho[n,128] @ w2[128,3] + b2 ; one wave per node
// ---------------------------------------------------------------------------
__global__ void head_kernel(const float* __restrict__ ho,
                            const float* __restrict__ w2,
                            const float* __restrict__ b2,
                            float* __restrict__ out, int n_rows)
{
  int row = blockIdx.x;
  int lane = threadIdx.x;
  if (row >= n_rows || lane >= 3) return;
  float s = b2[lane];
  for (int j = 0; j < 128; ++j) s += ho[(size_t)row * 128 + j] * w2[j * 3 + lane];
  out[(size_t)row * 3 + lane] = s;
}

// ===========================================================================
extern "C" void kernel_launch(void* const* d_in, const int* in_sizes, int n_in,
                              void* d_out, int out_size, void* d_ws, size_t ws_size,
                              hipStream_t stream)
{
  (void)in_sizes; (void)n_in; (void)out_size; (void)ws_size;
  // ---- inputs, flattened in setup_inputs() dict order -------------------
  const float* x       = (const float*)d_in[0];
  const int*   eidx    = (const int*)d_in[1];
  const float* eattr   = (const float*)d_in[2];
  // d_in[3] = batch (identity layout; unused)
  const float* node_w1 = (const float*)d_in[4];
  const float* node_b1 = (const float*)d_in[5];
  const float* node_w2 = (const float*)d_in[6];
  const float* node_b2 = (const float*)d_in[7];
  const float* edge_w1 = (const float*)d_in[8];
  const float* edge_b1 = (const float*)d_in[9];
  const float* edge_w2 = (const float*)d_in[10];
  const float* edge_b2 = (const float*)d_in[11];
  const float* g_wl    = (const float*)d_in[12];
  const float* g_bl    = (const float*)d_in[13];
  const float* g_wr    = (const float*)d_in[14];
  const float* g_br    = (const float*)d_in[15];
  const float* g_we    = (const float*)d_in[16];
  const float* g_att   = (const float*)d_in[17];
  const float* g_bo    = (const float*)d_in[18];
  const float* g_lng   = (const float*)d_in[19];
  const float* g_lnb   = (const float*)d_in[20];
  const float* t_wq    = (const float*)d_in[21];
  const float* t_wk    = (const float*)d_in[22];
  const float* t_wv    = (const float*)d_in[23];
  const float* t_wo    = (const float*)d_in[24];
  const float* t_bo    = (const float*)d_in[25];
  const float* t_ln1g  = (const float*)d_in[26];
  const float* t_ln1b  = (const float*)d_in[27];
  const float* t_ln2g  = (const float*)d_in[28];
  const float* t_ln2b  = (const float*)d_in[29];
  const float* t_ff1   = (const float*)d_in[30];
  const float* t_fb1   = (const float*)d_in[31];
  const float* t_ff2   = (const float*)d_in[32];
  const float* t_fb2   = (const float*)d_in[33];
  const float* out_lng = (const float*)d_in[34];
  const float* out_lnb = (const float*)d_in[35];
  const float* out_w1  = (const float*)d_in[36];
  const float* out_b1  = (const float*)d_in[37];
  const float* out_w2  = (const float*)d_in[38];
  const float* out_b2  = (const float*)d_in[39];
  const int* src = eidx;
  const int* dst = eidx + NEDGES;

  // ---- workspace layout --------------------------------------------------
  char* wp = (char*)d_ws;
  size_t off = 0;
  auto alloc = [&](size_t bytes) -> char* {
    char* p = wp + off;
    off += (bytes + 255) & ~(size_t)255;
    return p;
  };
  float* h    = (float*)alloc((size_t)NNODES * HIDC * 4);
  float* hn   = (float*)alloc((size_t)NNODES * HIDC * 4);
  float* bufA = (float*)alloc((size_t)NNODES * HIDC * 4);  // xl / q_f32 / f1 / ho
  float* bufB = (float*)alloc((size_t)NNODES * HIDC * 4);  // xr / k_f32
  float* bufC = (float*)alloc((size_t)NNODES * HIDC * 4);  // agg / v_f32
  float* bufD = (float*)alloc((size_t)NNODES * HIDC * 4);  // attn out
  unsigned short* hbf = (unsigned short*)alloc((size_t)NNODES * HIDC * 2);
  unsigned short* qbf = (unsigned short*)alloc((size_t)NNODES * HIDC * 2);
  unsigned short* kbf = (unsigned short*)alloc((size_t)NNODES * HIDC * 2);
  unsigned short* vpk = (unsigned short*)alloc((size_t)NB * NHEAD * (SEQ / 32) * 2 * 512 * 2);
  unsigned short* wpk = (unsigned short*)alloc((size_t)HIDC * HIDC * 2);  // packed weights
  float* ebuf   = (float*)alloc((size_t)NEDGES * EHC * 4);
  float* logitb = (float*)alloc((size_t)NEDGES * GHEAD * 4);
  float* exb    = (float*)alloc((size_t)NEDGES * GHEAD * 4);
  unsigned* menc = (unsigned*)alloc((size_t)NNODES * GHEAD * 4);
  float* den     = (float*)alloc((size_t)NNODES * GHEAD * 4);

  auto cvt = [&](const float* in, unsigned short* o, int n) {
    cvt_bf16_kernel<<<(n + 255) / 256, 256, 0, stream>>>(in, o, n);
  };
  auto packw = [&](const float* W, int N, int K) {
    int threads = (K >> 5) * (N >> 4) * 32;
    pack_w_kernel<<<(threads + 255) / 256, 256, 0, stream>>>(W, wpk, N, K);
  };
  auto gemm = [&](const unsigned short* Ab, const float* bias, const float* res,
                  float* C, int M, int N, int K, int act) {
    wmma_gemm_kernel<<<dim3(N / 64, M / 16), 32, 0, stream>>>(Ab, wpk, bias, res, C, M, N, K, act);
  };

  // ---- encoders ----------------------------------------------------------
  node_enc_kernel<<<NNODES / 8, 256, 0, stream>>>(x, node_w1, node_b1, node_w2, node_b2, h, NNODES);
  edge_enc_kernel<<<NEDGES / 8, 256, 0, stream>>>(eattr, edge_w1, edge_b1, edge_w2, edge_b2, ebuf, NEDGES);

  // ---- GAT layers --------------------------------------------------------
  for (int l = 0; l < NLAY; ++l) {
    const float* wl = g_wl + (size_t)l * HIDC * HIDC;
    const float* wr = g_wr + (size_t)l * HIDC * HIDC;
    const float* we = g_we + (size_t)l * EHC * HIDC;
    const float* bl = g_bl + (size_t)l * HIDC;
    const float* br = g_br + (size_t)l * HIDC;
    const float* bo = g_bo + (size_t)l * HIDC;
    const float* att = g_att + (size_t)l * GHEAD * GDH;
    const float* lng = g_lng + (size_t)l * HIDC;
    const float* lnb = g_lnb + (size_t)l * HIDC;

    cvt(h, hbf, NNODES * HIDC);
    packw(wl, HIDC, HIDC);
    gemm(hbf, bl, nullptr, bufA, NNODES, HIDC, HIDC, 0);   // xl
    packw(wr, HIDC, HIDC);
    gemm(hbf, br, nullptr, bufB, NNODES, HIDC, HIDC, 0);   // xr

    hipMemsetAsync(menc, 0, (size_t)NNODES * GHEAD * 4, stream);
    hipMemsetAsync(den,  0, (size_t)NNODES * GHEAD * 4, stream);
    hipMemsetAsync(bufC, 0, (size_t)NNODES * HIDC * 4, stream);

    gat_logit_kernel<<<NEDGES / 8, 256, 0, stream>>>(ebuf, we, bufA, bufB, src, dst, att,
                                                     logitb, menc, NEDGES);
    gat_softmax_kernel<<<(NEDGES * GHEAD + 255) / 256, 256, 0, stream>>>(logitb, menc, dst,
                                                                         exb, den, NEDGES * GHEAD);
    gat_scatter_kernel<<<NEDGES, 256, 0, stream>>>(bufB, exb, den, src, dst, bufC, NEDGES);

    // h = LN(h + agg + bo)
    ln_kernel<<<NNODES / 8, 256, 0, stream>>>(h, bufC, bo, lng, lnb, h, NNODES);
  }

  // ---- transformer layers ------------------------------------------------
  for (int l = 0; l < TLAY; ++l) {
    const float* wq = t_wq + (size_t)l * HIDC * HIDC;
    const float* wk = t_wk + (size_t)l * HIDC * HIDC;
    const float* wv = t_wv + (size_t)l * HIDC * HIDC;
    const float* wo = t_wo + (size_t)l * HIDC * HIDC;
    const float* bo = t_bo + (size_t)l * HIDC;
    const float* ff1 = t_ff1 + (size_t)l * HIDC * HIDC;
    const float* fb1 = t_fb1 + (size_t)l * HIDC;
    const float* ff2 = t_ff2 + (size_t)l * HIDC * HIDC;
    const float* fb2 = t_fb2 + (size_t)l * HIDC;

    // hn = LN1(h)
    ln_kernel<<<NNODES / 8, 256, 0, stream>>>(h, nullptr, nullptr,
                                              t_ln1g + (size_t)l * HIDC,
                                              t_ln1b + (size_t)l * HIDC, hn, NNODES);
    cvt(hn, hbf, NNODES * HIDC);
    packw(wq, HIDC, HIDC);
    gemm(hbf, nullptr, nullptr, bufA, NNODES, HIDC, HIDC, 0);
    packw(wk, HIDC, HIDC);
    gemm(hbf, nullptr, nullptr, bufB, NNODES, HIDC, HIDC, 0);
    packw(wv, HIDC, HIDC);
    gemm(hbf, nullptr, nullptr, bufC, NNODES, HIDC, HIDC, 0);
    cvt(bufA, qbf, NNODES * HIDC);
    cvt(bufB, kbf, NNODES * HIDC);
    pack_v_kernel<<<(NB * NHEAD * (SEQ / 32) * 2 * 32 + 255) / 256, 256, 0, stream>>>(bufC, vpk);

    attn_kernel<<<dim3(SEQ / 16, NHEAD, NB), 32, 0, stream>>>(qbf, kbf, vpk, bufD);

    // h = h + attn_out @ wo + bo
    cvt(bufD, hbf, NNODES * HIDC);
    packw(wo, HIDC, HIDC);
    gemm(hbf, bo, h, h, NNODES, HIDC, HIDC, 0);

    // hn = LN2(h); h = h + gelu(hn@ff1+fb1)@ff2 + fb2
    ln_kernel<<<NNODES / 8, 256, 0, stream>>>(h, nullptr, nullptr,
                                              t_ln2g + (size_t)l * HIDC,
                                              t_ln2b + (size_t)l * HIDC, hn, NNODES);
    cvt(hn, hbf, NNODES * HIDC);
    packw(ff1, HIDC, HIDC);
    gemm(hbf, fb1, nullptr, bufA, NNODES, HIDC, HIDC, 1);
    cvt(bufA, hbf, NNODES * HIDC);
    packw(ff2, HIDC, HIDC);
    gemm(hbf, fb2, h, h, NNODES, HIDC, HIDC, 0);
  }

  // ---- output head -------------------------------------------------------
  ln_kernel<<<NNODES / 8, 256, 0, stream>>>(h, nullptr, nullptr, out_lng, out_lnb, hn, NNODES);
  cvt(hn, hbf, NNODES * HIDC);
  packw(out_w1, 128, HIDC);
  gemm(hbf, out_b1, nullptr, bufA, NNODES, 128, HIDC, 1);   // [N,128] gelu
  head_kernel<<<NNODES, 32, 0, stream>>>(bufA, out_w2, out_b2, (float*)d_out, NNODES);
}
